// BahdanauAttention_6682969112620
// MI455X (gfx1250) — matrix operand order repaired and checked by reading, not compiled, not run on hardware
//
#include <hip/hip_runtime.h>

// ---------------------------------------------------------------------------
// Types for CDNA5 WMMA
// ---------------------------------------------------------------------------
typedef __attribute__((ext_vector_type(16))) __bf16 v16bf;
typedef __attribute__((ext_vector_type(8)))  float  v8f;
typedef __attribute__((ext_vector_type(4)))  int    v4i;

union FragBF16 {
    v16bf v;
    uint4 q[2];
};

#define AS1 __attribute__((address_space(1)))
#define AS3 __attribute__((address_space(3)))

#if defined(__AMDGCN__) && __has_builtin(__builtin_amdgcn_global_load_async_to_lds_b128)
#define HAVE_ASYNC_LDS 1
#else
#define HAVE_ASYNC_LDS 0
#endif

// 16-byte global -> LDS copy: async DMA path on gfx1250, sync fallback else.
// Builtin signature (from hipcc diagnostic): (AS1 v4i* src, AS3 v4i* dst, imm, imm).
static __device__ __forceinline__ void cp16_g2l(void* lds_dst, const void* gsrc) {
#if HAVE_ASYNC_LDS
    __builtin_amdgcn_global_load_async_to_lds_b128(
        (AS1 v4i*)gsrc, (AS3 v4i*)lds_dst, 0, 0);
#else
    *(uint4*)lds_dst = *(const uint4*)gsrc;
#endif
}

static __device__ __forceinline__ void wait_async_copies() {
#if HAVE_ASYNC_LDS
#if __has_builtin(__builtin_amdgcn_s_wait_asynccnt)
    __builtin_amdgcn_s_wait_asynccnt(0);
#else
    asm volatile("s_wait_asynccnt 0" ::: "memory");
#endif
#endif
}

// float -> bf16 (round-to-nearest-even), raw u16 bit layout.
static __device__ __forceinline__ unsigned short f2bf(float f) {
    unsigned int u = __float_as_uint(f);
    unsigned int r = (u + 0x7FFFu + ((u >> 16) & 1u)) >> 16;
    return (unsigned short)r;
}

static __device__ __forceinline__ float fast_tanh(float x) {
#if __has_builtin(__builtin_amdgcn_tanhf)
    return __builtin_amdgcn_tanhf(x);          // v_tanh_f32 on gfx1250
#else
    float e = __expf(2.0f * x);                // v_exp_f32
    return 1.0f - 2.0f * __builtin_amdgcn_rcpf(e + 1.0f);
#endif
}

// ---------------------------------------------------------------------------
// Prep kernel 1: elementwise f32 -> bf16 (packed stores).
// ---------------------------------------------------------------------------
__global__ __launch_bounds__(256)
void convert_f32_bf16(const float* __restrict__ src, unsigned short* __restrict__ dst,
                      int n4 /* number of float4 groups */)
{
    int i = blockIdx.x * 256 + threadIdx.x;
    if (i >= n4) return;
    float4 v = *(const float4*)(src + (long long)i * 4);
    uint2 p;
    p.x = (unsigned int)f2bf(v.x) | ((unsigned int)f2bf(v.y) << 16);
    p.y = (unsigned int)f2bf(v.z) | ((unsigned int)f2bf(v.w) << 16);
    *(uint2*)(dst + (long long)i * 4) = p;
}

// ---------------------------------------------------------------------------
// Prep kernel 2: tiled transpose, f32 [R][C] -> bf16 [C][R], batched on z.
// ---------------------------------------------------------------------------
__global__ __launch_bounds__(256)
void transpose_f32_to_bf16(const float* __restrict__ src, int R, int C,
                           long long sstride,
                           unsigned short* __restrict__ dst, long long dstride)
{
    __shared__ float tile[32][33];
    const float* s = src + (long long)blockIdx.z * sstride;
    unsigned short* d = dst + (long long)blockIdx.z * dstride;
    const int c0 = blockIdx.x * 32, r0 = blockIdx.y * 32;
    const int tx = threadIdx.x & 31, ty = threadIdx.x >> 5;   // 32 x 8
#pragma unroll
    for (int i = 0; i < 32; i += 8)
        tile[ty + i][tx] = s[(long long)(r0 + ty + i) * C + c0 + tx];
    __syncthreads();
#pragma unroll
    for (int i = 0; i < 32; i += 8)
        d[(long long)(c0 + ty + i) * R + r0 + tx] = f2bf(tile[tx][ty + i]);
}

// ---------------------------------------------------------------------------
// GEMM: C[b] = A[b] (MxK) * B[b] (KxN); A is [M][K] bf16 row-major,
// BT is [N][K] bf16 row-major (pre-transposed B). C is f32 [M][N].
// Block 256 threads (8 waves); block tile 128x128, BK=32; double-buffered LDS
// filled via async global->LDS b128 copies; 8 back-to-back WMMAs per k-step.
// ---------------------------------------------------------------------------
#define BM 128
#define BN 128
#define BK 32
#define LSTR 48   // bf16 elems per LDS row: 32 used + 16 pad (96 B, 16B-mult)

__global__ __launch_bounds__(256)
void wmma_gemm_bf16(const unsigned short* __restrict__ Ag, int lda, long long strideA,
                    const unsigned short* __restrict__ BTg, int ldbt, long long strideBT,
                    float* __restrict__ Cg, int ldc, long long strideC,
                    int M, int N, int K)
{
    __shared__ __align__(16) unsigned short Als[2][BM * LSTR]; // [buf][row][k]
    __shared__ __align__(16) unsigned short Bls[2][BN * LSTR]; // [buf][col][k]

    const int tid  = threadIdx.x;
    const int lane = tid & 31;
    const int wave = tid >> 5;

    const int nColBlocks = N / BN;
    const int rowBlk = (blockIdx.x / nColBlocks) * BM;
    const int colBlk = (blockIdx.x % nColBlocks) * BN;

    const unsigned short* A  = Ag  + (long long)blockIdx.y * strideA;
    const unsigned short* BT = BTg + (long long)blockIdx.y * strideBT;
    float*                C  = Cg  + (long long)blockIdx.y * strideC;

    v8f acc[8];
#pragma unroll
    for (int i = 0; i < 8; ++i)
#pragma unroll
        for (int j = 0; j < 8; ++j) acc[i][j] = 0.0f;

    const int laneM  = lane & 15;
    const int khalfA = (lane < 16) ? 0 : 8;   // A frag K run base
    const int kbaseB = (lane < 16) ? 0 : 8;   // B frag K base

    const int nk = K / BK;

    // Issue the 16B-chunk copies for k-step `ks` into buffer `buf`.
    // Tile = 128 rows x 32 bf16 = 512 chunks of 16B; 2 chunks/thread/operand.
    auto issue_tile = [&](int ks, int buf) {
        const int k0 = ks * BK;
#pragma unroll
        for (int h = 0; h < 2; ++h) {
            const int c  = tid + h * 256;
            const int r  = c >> 2;
            const int kc = (c & 3) * 8;      // ushort offset, 16B granules
            cp16_g2l(&Als[buf][r * LSTR + kc],
                     A + (long long)(rowBlk + r) * lda + k0 + kc);
            cp16_g2l(&Bls[buf][r * LSTR + kc],
                     BT + (long long)(colBlk + r) * ldbt + k0 + kc);
        }
    };

    issue_tile(0, 0);

    for (int ks = 0; ks < nk; ++ks) {
        wait_async_copies();     // my chunks for tile `ks` have landed in LDS
        __syncthreads();         // everyone's chunks landed; prev buffer free

        const int buf = ks & 1;
        if (ks + 1 < nk) issue_tile(ks + 1, buf ^ 1);
        if (ks + 2 < nk)  // hint the tile after next toward L2
            __builtin_prefetch(A + (long long)(rowBlk + (tid >> 2)) * lda +
                                   (ks + 2) * BK + (tid & 3) * 8, 0, 0);

        // A fragment: 16x32 bf16. Lane L<16: row M=L, K {0..7,16..23};
        // lane L+16: row M=L, K {8..15,24..31}.
        FragBF16 af;
        const int arow = wave * 16 + laneM;
        af.q[0] = *(const uint4*)&Als[buf][arow * LSTR + khalfA];
        af.q[1] = *(const uint4*)&Als[buf][arow * LSTR + khalfA + 16];

        // Preload ALL 8 B fragments, then 8 back-to-back WMMAs.
        FragBF16 bfr[8];
#pragma unroll
        for (int ct = 0; ct < 8; ++ct) {
            const int bcol = ct * 16 + laneM;
            bfr[ct].q[0] = *(const uint4*)&Bls[buf][bcol * LSTR + kbaseB];
            bfr[ct].q[1] = *(const uint4*)&Bls[buf][bcol * LSTR + kbaseB + 16];
        }
#pragma unroll
        for (int ct = 0; ct < 8; ++ct)
            acc[ct] = __builtin_amdgcn_wmma_f32_16x16x32_bf16(
                false, af.v, false, bfr[ct].v, (short)0, acc[ct], false, false);
    }

    // C/D layout: VGPR r -> M = r (lanes 0-15) or 8+r (lanes 16-31), N = lane%16.
    const int rbase = rowBlk + wave * 16 + ((lane < 16) ? 0 : 8);
#pragma unroll
    for (int ct = 0; ct < 8; ++ct) {
        const int col = colBlk + ct * 16 + laneM;
#pragma unroll
        for (int r = 0; r < 8; ++r)
            C[(long long)(rbase + r) * ldc + col] = acc[ct][r];
    }
}

// ---------------------------------------------------------------------------
// Additive scores + masked softmax (transcendental-bound stage).
// Block = (batch b, 8 query rows). 256 threads: tid&127 = key s, tid>>7 = u-half.
// ---------------------------------------------------------------------------
#define TT 8

__global__ __launch_bounds__(256)
void scores_softmax(const float* __restrict__ wq, const float* __restrict__ wk,
                    const float* __restrict__ scale,
                    const unsigned char* __restrict__ mask,
                    float* __restrict__ attn,
                    unsigned short* __restrict__ attn_bf)
{
    __shared__ __align__(16) float wqs[TT][1024];   // 32 KB
    __shared__ __align__(16) float scl[1024];       //  4 KB
    __shared__ float red[256][TT];                  //  8 KB
    __shared__ float rowstat[TT];

    const int tid = threadIdx.x;
    const int b   = blockIdx.x >> 4;
    const int t0  = (blockIdx.x & 15) * TT;

    for (int i = tid; i < TT * 256; i += 256) {       // 256 float4 per row
        int tt = i >> 8;
        int u4 = (i & 255) * 4;
        *(float4*)&wqs[tt][u4] =
            *(const float4*)&wq[((long long)(b * 128 + t0 + tt)) * 1024 + u4];
    }
    {
        int u4 = tid * 4;
        *(float4*)&scl[u4] = *(const float4*)&scale[u4];
    }
    __syncthreads();

    const int s  = tid & 127;
    const int uh = tid >> 7;

    float acc[TT];
#pragma unroll
    for (int tt = 0; tt < TT; ++tt) acc[tt] = 0.0f;

    const float* wkrow = &wk[((long long)(b * 128 + s)) * 1024 + uh * 512];
    const int ub = uh * 512;
    for (int u = 0; u < 512; u += 4) {
        float4 wk4 = *(const float4*)(wkrow + u);
        float4 sc4 = *(const float4*)&scl[ub + u];
#pragma unroll
        for (int tt = 0; tt < TT; ++tt) {
            float4 wq4 = *(const float4*)&wqs[tt][ub + u];
            acc[tt] += sc4.x * fast_tanh(wq4.x + wk4.x);
            acc[tt] += sc4.y * fast_tanh(wq4.y + wk4.y);
            acc[tt] += sc4.z * fast_tanh(wq4.z + wk4.z);
            acc[tt] += sc4.w * fast_tanh(wq4.w + wk4.w);
        }
    }

#pragma unroll
    for (int tt = 0; tt < TT; ++tt) red[tid][tt] = acc[tt];
    __syncthreads();

    float sc[TT];
    if (uh == 0) {
        const bool mk = mask[b * 128 + s] != 0;
#pragma unroll
        for (int tt = 0; tt < TT; ++tt) {
            float v = acc[tt] + red[tid + 128][tt];
            sc[tt] = mk ? v : -1e9f;
        }
#pragma unroll
        for (int tt = 0; tt < TT; ++tt) red[s][tt] = sc[tt];
    }
    __syncthreads();

    if (tid < TT) {                    // per-row max over 128 keys
        float m = -3.4e38f;
        for (int ss = 0; ss < 128; ++ss) m = fmaxf(m, red[ss][tid]);
        rowstat[tid] = m;
    }
    __syncthreads();

    float e[TT];
    if (uh == 0) {
#pragma unroll
        for (int tt = 0; tt < TT; ++tt) {
            e[tt] = __expf(sc[tt] - rowstat[tt]);   // v_exp_f32
            red[s][tt] = e[tt];
        }
    }
    __syncthreads();

    if (tid < TT) {                    // per-row sum + reciprocal
        float sum = 0.0f;
        for (int ss = 0; ss < 128; ++ss) sum += red[ss][tid];
        rowstat[tid] = __builtin_amdgcn_rcpf(sum);
    }
    __syncthreads();

    if (uh == 0) {
#pragma unroll
        for (int tt = 0; tt < TT; ++tt) {
            float a = e[tt] * rowstat[tt];
            long long idx = ((long long)(b * 128 + t0 + tt)) * 128 + s;
            attn[idx] = a;             // f32 output (second tuple element)
            attn_bf[idx] = f2bf(a);    // bf16 operand for the context GEMM
        }
    }
}

// ---------------------------------------------------------------------------
// Launcher. Inputs (setup_inputs order):
//   0 query [8,128,512] f32   1 value [8,128,512] f32   2 mask [8,128] bool
//   3 W1 [512,1024] f32       4 W2 [512,1024] f32       5 scale [1024] f32
// Output: context [8,128,512] f32 ++ attn [8,128,128] f32.
// ---------------------------------------------------------------------------
extern "C" void kernel_launch(void* const* d_in, const int* in_sizes, int n_in,
                              void* d_out, int out_size, void* d_ws, size_t ws_size,
                              hipStream_t stream) {
    (void)in_sizes; (void)n_in; (void)out_size; (void)ws_size;

    const float*         q     = (const float*)d_in[0];
    const float*         v     = (const float*)d_in[1];
    const unsigned char* mask  = (const unsigned char*)d_in[2];
    const float*         W1    = (const float*)d_in[3];
    const float*         W2    = (const float*)d_in[4];
    const float*         scale = (const float*)d_in[5];

    float* ctx  = (float*)d_out;                     // [8,128,512]
    float* attn = ctx + (long long)8 * 128 * 512;    // [8,128,128]

    // Workspace layout
    float* wq = (float*)d_ws;                        // [1024][1024] f32
    float* wk = wq + 1048576;                        // [1024][1024] f32
    unsigned short* qbf    = (unsigned short*)(wk + 1048576); // [1024][512]
    unsigned short* vbf    = qbf + 524288;                    // [1024][512]
    unsigned short* W1T    = vbf + 524288;                    // [1024][512]
    unsigned short* W2T    = W1T + 524288;                    // [1024][512]
    unsigned short* vT     = W2T + 524288;                    // [8][512][128]
    unsigned short* attnbf = vT + 524288;                     // [8][128][128]

    // --- prep: bf16 conversions / transposes (L2-resident, cold path) ---
    convert_f32_bf16<<<dim3(512), 256, 0, stream>>>(q, qbf, 131072);
    convert_f32_bf16<<<dim3(512), 256, 0, stream>>>(v, vbf, 131072);
    transpose_f32_to_bf16<<<dim3(32, 16, 1), 256, 0, stream>>>(
        W1, 512, 1024, 0LL, W1T, 0LL);
    transpose_f32_to_bf16<<<dim3(32, 16, 1), 256, 0, stream>>>(
        W2, 512, 1024, 0LL, W2T, 0LL);
    transpose_f32_to_bf16<<<dim3(16, 4, 8), 256, 0, stream>>>(
        v, 128, 512, (long long)128 * 512, vT, (long long)512 * 128);

    // --- projections: wq = q@W1, wk = v@W2 (M=N=1024, K=512) ---
    wmma_gemm_bf16<<<dim3(64, 1), 256, 0, stream>>>(
        qbf, 512, 0LL, W1T, 512, 0LL, wq, 1024, 0LL, 1024, 1024, 512);
    wmma_gemm_bf16<<<dim3(64, 1), 256, 0, stream>>>(
        vbf, 512, 0LL, W2T, 512, 0LL, wk, 1024, 0LL, 1024, 1024, 512);

    // --- additive scores + masked softmax ---
    scores_softmax<<<dim3(128), 256, 0, stream>>>(wq, wk, scale, mask, attn, attnbf);

    // --- context = attn @ value (batched, M=128, N=512, K=128) ---
    wmma_gemm_bf16<<<dim3(4, 8), 256, 0, stream>>>(
        attnbf, 128, (long long)128 * 128,
        vT,     128, (long long)512 * 128,
        ctx,    512, (long long)128 * 512,
        128, 512, 128);
}
